// SAINT_23699629539722
// MI455X (gfx1250) — matrix operand order repaired and checked by reading, not compiled, not run on hardware
//
#include <hip/hip_runtime.h>

typedef __attribute__((ext_vector_type(16))) _Float16 v16h;
typedef __attribute__((ext_vector_type(8)))  float    v8f;
typedef __attribute__((ext_vector_type(4)))  unsigned u32x4;
typedef __attribute__((ext_vector_type(4)))  float    f32x4;

union AFrag {
  u32x4    q[2];
  unsigned u[8];
  _Float16 h[16];
  v16h     v;
};

union H8 {
  _Float16 h[8];
  u32x4    q;
};

// ---------------------------------------------------------------------------
// Pack a 128x128 row-major fp32 weight W[k][n] into the WMMA f16 B-fragment
// layout for 16x16x32: per (ktile, ntile) a 32-lane x 8-dword block where
// lane holds col n=nt*16+(lane&15); half h=lane>>4, dword i<4 holds K=8h+2i,+1,
// dword i>=4 holds K=16+8h+2(i-4),+1.  Four matrices packed back to back (8192 dwords each).
// ---------------------------------------------------------------------------
__global__ void pack_weights_k(const float* __restrict__ W0, const float* __restrict__ W1,
                               const float* __restrict__ W2, const float* __restrict__ W3,
                               unsigned* __restrict__ Bp) {
  int t = blockIdx.x * blockDim.x + threadIdx.x;
  if (t >= 4 * 8192) return;
  int m = t >> 13, r = t & 8191;
  int i = r & 7, lane = (r >> 3) & 31, ktnt = r >> 8;
  int nt = ktnt & 7, kt = ktnt >> 3;
  const float* W = (m == 0) ? W0 : (m == 1) ? W1 : (m == 2) ? W2 : W3;
  int n = nt * 16 + (lane & 15);
  int k = kt * 32 + (lane >> 4) * 8 + (i >> 2) * 16 + 2 * (i & 3);
  union { _Float16 hh[2]; unsigned u; } cv;
  cv.hh[0] = (_Float16)W[k * 128 + n];
  cv.hh[1] = (_Float16)W[(k + 1) * 128 + n];
  Bp[t] = cv.u;
}

// fp32 -> f16 row-major copy, 8 elements per thread
__global__ void cvt_f32_f16_k(const float* __restrict__ in, unsigned short* __restrict__ out,
                              long long n8) {
  long long t = blockIdx.x * (long long)blockDim.x + threadIdx.x;
  if (t >= n8) return;
  const f32x4* p = (const f32x4*)(in + t * 8);
  f32x4 a = p[0], b = p[1];
  H8 o;
  o.h[0] = (_Float16)a[0]; o.h[1] = (_Float16)a[1]; o.h[2] = (_Float16)a[2]; o.h[3] = (_Float16)a[3];
  o.h[4] = (_Float16)b[0]; o.h[5] = (_Float16)b[1]; o.h[6] = (_Float16)b[2]; o.h[7] = (_Float16)b[3];
  *(u32x4*)(out + t * 8) = o.q;
}

// segment_sum(x[src], dst): one thread per (edge, 4-feature chunk).
// x (51 MB) is L2-resident, so gather + fp32 atomics stay on-die.
__global__ void scatter_add_k(const int* __restrict__ ei, const float* __restrict__ xin,
                              float* __restrict__ agg, int E) {
  long long t = blockIdx.x * (long long)blockDim.x + threadIdx.x;
  if (t >= (long long)E * 32) return;
  int e = (int)(t >> 5);
  int c = ((int)t & 31) * 4;
  int src = ei[e];
  int dst = ei[E + e];
  f32x4 v = *(const f32x4*)(xin + (size_t)src * 128 + c);
  float* o = agg + (size_t)dst * 128 + c;
  unsafeAtomicAdd(o + 0, v[0]);
  unsafeAtomicAdd(o + 1, v[1]);
  unsafeAtomicAdd(o + 2, v[2]);
  unsafeAtomicAdd(o + 3, v[3]);
}

// ---------------------------------------------------------------------------
// Fused GraphConv GEMM: out = relu(agg @ W_rel + bias + x @ W_root)
// One wave per 16-row tile, 8 column tiles (128 cols), K=128 in 4 steps of 32.
// 64 v_wmma_f32_16x16x32_f16 per wave, f32 accumulation.
// A-fragment (16-bit 16x32): lane half h = lane>>4 holds K=[8h..8h+7] in
// VGPR0-3 and K=[16+8h..16+8h+7] in VGPR4-7, row M = lane&15.
// ---------------------------------------------------------------------------
__global__ void __launch_bounds__(256)
gconv_gemm_k(const float* __restrict__ aggF, const unsigned short* __restrict__ xh,
             const unsigned* __restrict__ BpRel, const unsigned* __restrict__ BpRoot,
             const float* __restrict__ bias, float* __restrict__ outF,
             unsigned short* __restrict__ outH, int N, int ntiles) {
  int tile = blockIdx.x * blockDim.y + threadIdx.y;
  if (tile >= ntiles) return;           // whole wave exits together: EXEC stays full
  int lane = threadIdx.x;
  int mcol = lane & 15;
  int hsel = lane >> 4;
  int row0 = tile * 16;
  int r = row0 + mcol;
  if (r >= N) r = N - 1;                // clamp loads, EXEC stays all-ones

  v8f acc[8];
  #pragma unroll
  for (int nt = 0; nt < 8; ++nt)
    #pragma unroll
    for (int j = 0; j < 8; ++j) acc[nt][j] = 0.f;

  #pragma unroll
  for (int kt = 0; kt < 4; ++kt) {
    const int k0 = kt * 32;
    AFrag aX, aG;
    {
      // root path A: f16 straight from memory (two b128 loads, K and K+16)
      const u32x4* xp = (const u32x4*)(xh + (size_t)r * 128 + k0 + hsel * 8);
      aX.q[0] = xp[0];
      aX.q[1] = xp[2];
      // aggregate path A: fp32 -> f16 in-register
      const f32x4* gp = (const f32x4*)(aggF + (size_t)r * 128 + k0 + hsel * 8);
      f32x4 f0 = gp[0], f1 = gp[1], f2 = gp[4], f3 = gp[5];
      #pragma unroll
      for (int j = 0; j < 4; ++j) {
        aG.h[j]      = (_Float16)f0[j];
        aG.h[4 + j]  = (_Float16)f1[j];
        aG.h[8 + j]  = (_Float16)f2[j];
        aG.h[12 + j] = (_Float16)f3[j];
      }
    }
    #pragma unroll
    for (int nt = 0; nt < 8; ++nt) {
      const unsigned base = ((unsigned)(kt * 8 + nt) * 32 + (unsigned)lane) << 3;
      const u32x4* bp = (const u32x4*)(BpRel + base);
      AFrag bR; bR.q[0] = bp[0]; bR.q[1] = bp[1];
      acc[nt] = __builtin_amdgcn_wmma_f32_16x16x32_f16(
          false, aG.v, false, bR.v, (short)0, acc[nt], false, false);
      const u32x4* bp2 = (const u32x4*)(BpRoot + base);
      AFrag bT; bT.q[0] = bp2[0]; bT.q[1] = bp2[1];
      acc[nt] = __builtin_amdgcn_wmma_f32_16x16x32_f16(
          false, aX.v, false, bT.v, (short)0, acc[nt], false, false);
    }
  }

  // C/D layout: col = lane&15, row = row0 + 8*(lane>>4) + vgpr. Fuse bias+ReLU,
  // emit fp32 (for next scatter / head) and optional f16 (next GEMM's A).
  #pragma unroll
  for (int nt = 0; nt < 8; ++nt) {
    int col = nt * 16 + mcol;
    float bc = bias[col];
    #pragma unroll
    for (int i = 0; i < 8; ++i) {
      int row = row0 + hsel * 8 + i;
      if (row < N) {
        float val = acc[nt][i] + bc;
        val = val > 0.f ? val : 0.f;
        outF[(size_t)row * 128 + col] = val;
        if (outH) {
          union { _Float16 hh; unsigned short us; } cv;
          cv.hh = (_Float16)val;
          outH[(size_t)row * 128 + col] = cv.us;
        }
      }
    }
  }
}

// Head: logits = [x1,x2] @ W_lin + b_lin, then log_softmax.  One block per node.
__global__ void __launch_bounds__(64)
head_k(const float* __restrict__ x1, const float* __restrict__ x2,
       const float* __restrict__ Wl, const float* __restrict__ bl,
       float* __restrict__ out, int N) {
  __shared__ float hbuf[256];
  __shared__ float lbuf[40];
  __shared__ float stats[2];
  int row = blockIdx.x;
  if (row >= N) return;
  int tid = threadIdx.x;
  const float* s1 = x1 + (size_t)row * 128;
  const float* s2 = x2 + (size_t)row * 128;
  hbuf[tid * 2]           = s1[tid * 2];
  hbuf[tid * 2 + 1]       = s1[tid * 2 + 1];
  hbuf[128 + tid * 2]     = s2[tid * 2];
  hbuf[128 + tid * 2 + 1] = s2[tid * 2 + 1];
  __syncthreads();
  if (tid < 40) {
    float acc = bl[tid];
    #pragma unroll 4
    for (int k = 0; k < 256; ++k) acc += hbuf[k] * Wl[k * 40 + tid];
    lbuf[tid] = acc;
  }
  __syncthreads();
  if (tid == 0) {
    float mx = lbuf[0];
    for (int c = 1; c < 40; ++c) mx = fmaxf(mx, lbuf[c]);
    float s = 0.f;
    for (int c = 0; c < 40; ++c) s += __expf(lbuf[c] - mx);
    stats[0] = mx;
    stats[1] = __logf(s);
  }
  __syncthreads();
  if (tid < 40) out[(size_t)row * 40 + tid] = lbuf[tid] - stats[0] - stats[1];
}

extern "C" void kernel_launch(void* const* d_in, const int* in_sizes, int n_in,
                              void* d_out, int out_size, void* d_ws, size_t ws_size,
                              hipStream_t stream) {
  const float* x      = (const float*)d_in[0];
  const int*   ei     = (const int*)d_in[1];
  const float* Wrel1  = (const float*)d_in[2];
  const float* brel1  = (const float*)d_in[3];
  const float* Wroot1 = (const float*)d_in[4];
  const float* Wrel2  = (const float*)d_in[5];
  const float* brel2  = (const float*)d_in[6];
  const float* Wroot2 = (const float*)d_in[7];
  const float* Wlin   = (const float*)d_in[8];
  const float* blin   = (const float*)d_in[9];
  float* out = (float*)d_out;

  const int N = in_sizes[0] / 128;   // 100000
  const int E = in_sizes[1] / 2;     // 1600000
  const int ntiles = (N + 15) / 16;

  // workspace layout (bytes), all regions 256B-aligned by construction
  char* ws = (char*)d_ws;
  size_t off = 0;
  float* agg = (float*)(ws + off);            off += (size_t)N * 128 * 4;
  float* x1  = (float*)(ws + off);            off += (size_t)N * 128 * 4;
  float* x2  = (float*)(ws + off);            off += (size_t)N * 128 * 4;
  unsigned short* xh  = (unsigned short*)(ws + off); off += (size_t)N * 128 * 2;
  unsigned short* x1h = (unsigned short*)(ws + off); off += (size_t)N * 128 * 2;
  unsigned* Bp = (unsigned*)(ws + off);       off += (size_t)4 * 8192 * 4;
  (void)ws_size; (void)n_in; (void)out_size;

  // 1) pack weights into WMMA B-fragment layout
  pack_weights_k<<<(4 * 8192 + 255) / 256, 256, 0, stream>>>(Wrel1, Wroot1, Wrel2, Wroot2, Bp);

  // 2) x -> f16
  long long n8 = (long long)N * 128 / 8;
  cvt_f32_f16_k<<<(unsigned)((n8 + 255) / 256), 256, 0, stream>>>(x, xh, n8);

  dim3 gemm_grid((ntiles + 7) / 8), gemm_block(32, 8);
  long long st = (long long)E * 32;
  unsigned sblocks = (unsigned)((st + 255) / 256);

  // layer 1
  hipMemsetAsync(agg, 0, (size_t)N * 128 * 4, stream);
  scatter_add_k<<<sblocks, 256, 0, stream>>>(ei, x, agg, E);
  gconv_gemm_k<<<gemm_grid, gemm_block, 0, stream>>>(agg, xh, Bp, Bp + 8192,
                                                     brel1, x1, x1h, N, ntiles);
  // layer 2
  hipMemsetAsync(agg, 0, (size_t)N * 128 * 4, stream);
  scatter_add_k<<<sblocks, 256, 0, stream>>>(ei, x1, agg, E);
  gconv_gemm_k<<<gemm_grid, gemm_block, 0, stream>>>(agg, x1h, Bp + 16384, Bp + 24576,
                                                     brel2, x2, nullptr, N, ntiles);
  // head + log_softmax
  head_k<<<N, 64, 0, stream>>>(x1, x2, Wlin, blin, out, N);
}
